// SSDLoss_15281493639907
// MI455X (gfx1250) — compile-verified, matrix operand order checked by reading, not run on hardware
//
#include <hip/hip_runtime.h>
#include <math.h>

typedef __attribute__((ext_vector_type(16))) _Float16 v16h;
typedef __attribute__((ext_vector_type(8)))  float    v8f;

#define B_  64
#define D_  24564
#define C_  81
#define STRIDE_ 85
#define WAVES_  4
#define ROWS_PER_BLOCK (WAVES_ * 16)   /* 64 */
#define NCHUNK ((D_ + ROWS_PER_BLOCK - 1) / ROWS_PER_BLOCK)  /* 384 */
#define NPART  (B_ * NCHUNK)
#define NEG_FILL_ (-1e30f)

#define HAS_ASYNC_LDS __has_builtin(__builtin_amdgcn_global_load_async_to_lds_b32)
#define HAS_WAIT_ASYNC __has_builtin(__builtin_amdgcn_s_wait_asynccnt)

typedef __attribute__((address_space(1))) int g1int;
typedef __attribute__((address_space(3))) int l3int;

__device__ __forceinline__ unsigned mapF(float f) {
  unsigned u = __float_as_uint(f);
  return (u & 0x80000000u) ? ~u : (u | 0x80000000u);
}
__device__ __forceinline__ float unmapF(unsigned u) {
  return (u & 0x80000000u) ? __uint_as_float(u & 0x7FFFFFFFu)
                           : __uint_as_float(~u);
}

// Stage one float from global to LDS: async (CDNA5 ASYNCcnt path) if available.
__device__ __forceinline__ void stage1(const float* __restrict__ g, float* l) {
#if HAS_ASYNC_LDS
  __builtin_amdgcn_global_load_async_to_lds_b32((g1int*)g, (l3int*)l, 0, 0);
#else
  *l = *g;
#endif
}

__device__ __forceinline__ void stage_fence() {
#if HAS_ASYNC_LDS && HAS_WAIT_ASYNC
  __builtin_amdgcn_s_wait_asynccnt(0);
#endif
  __syncthreads();
}

// ---------------------------------------------------------------------------
// Kernel 1: stage 16-row blocks to LDS (coalesced), softmax-CE row sums via
// WMMA (exp values x all-ones B), smooth-L1 loc, deterministic block partials.
// ---------------------------------------------------------------------------
__global__ void __launch_bounds__(128) ssd_row_kernel(
    const float* __restrict__ predicts, const float* __restrict__ gt_loc,
    const float* __restrict__ gt_conf, const unsigned char* __restrict__ pos,
    float* __restrict__ nloss, float* __restrict__ partP,
    float* __restrict__ partLoc, int* __restrict__ partN)
{
  __shared__ float ldsP[WAVES_][16 * STRIDE_];  // 4 x 1360 floats
  __shared__ float ldsC[WAVES_][16 * C_];       // 4 x 1296 floats

  const int tid  = threadIdx.x;
  const int lane = tid & 31;
  const int wave = tid >> 5;
  const int bIdx = blockIdx.y;
  const int chnk = blockIdx.x;
  const int isHigh = lane >> 4;
  const int M = lane & 15;
  const int d0 = chnk * ROWS_PER_BLOCK + wave * 16;
  const int d  = d0 + M;
  const bool valid = (d < D_);

  // ---- coalesced staging: lane-consecutive dwords, 1 cacheline / vmem op ----
  const int vrows = (D_ - d0) >= 16 ? 16 : ((D_ - d0) > 0 ? (D_ - d0) : 0);
  const int nvP = vrows * STRIDE_;
  const int nvC = vrows * C_;
  const size_t gP = ((size_t)bIdx * D_ + (size_t)d0) * STRIDE_;
  const size_t gC = ((size_t)bIdx * D_ + (size_t)d0) * C_;
  float* lp = &ldsP[wave][0];
  float* lq = &ldsC[wave][0];
  for (int i = lane; i < nvP; i += 32) stage1(predicts + gP + i, lp + i);
  for (int i = lane; i < nvC; i += 32) stage1(gt_conf + gC + i, lq + i);
  stage_fence();

  // ---- per-lane gather from LDS (lane pair m / m+16 covers one row) ----
  float lv[48];
  float mx  = -INFINITY;
  float dot = 0.0f;
#pragma unroll
  for (int s = 0; s < 48; ++s) {
    const int within = s & 15;
    const int kk  = within + ((within >= 8) ? 8 : 0) + (isHigh ? 8 : 0);
    const int cls = (s >> 4) * 32 + kk;
    float v = -INFINITY;
    if (valid && cls < C_) {
      v = lp[M * STRIDE_ + 4 + cls];
      dot += lq[M * C_ + cls] * v;   // one-hot dot -> logit[label]
      mx = fmaxf(mx, v);
    }
    lv[s] = v;
  }
  mx  = fmaxf(mx, __shfl_xor(mx, 16, 32));
  dot += __shfl_xor(dot, 16, 32);
  if (!valid) mx = 0.0f;   // keep exp(-inf - mx) == 0 (avoid NaN)

  // ---- pack exp(x - max) into three 16x32 f16 A tiles ----
  v16h A0, A1, A2, ones;
#pragma unroll
  for (int i = 0; i < 16; ++i) A0[i] = (_Float16)__expf(lv[i]      - mx);
#pragma unroll
  for (int i = 0; i < 16; ++i) A1[i] = (_Float16)__expf(lv[16 + i] - mx);
#pragma unroll
  for (int i = 0; i < 16; ++i) A2[i] = (_Float16)__expf(lv[32 + i] - mx);
#pragma unroll
  for (int i = 0; i < 16; ++i) ones[i] = (_Float16)1.0f;

  // Row sums via WMMA: C[M][N] = sum_K A[M][K] * 1 (every column = row sum)
  v8f c = {};
  c = __builtin_amdgcn_wmma_f32_16x16x32_f16(false, A0, false, ones, (short)0, c, false, false);
  c = __builtin_amdgcn_wmma_f32_16x16x32_f16(false, A1, false, ones, (short)0, c, false, false);
  c = __builtin_amdgcn_wmma_f32_16x16x32_f16(false, A2, false, ones, (short)0, c, false, false);

  // Extract S per row (C layout: lanes 0-15 vgpr r -> M=r; lanes 16-31 -> M=8+r)
  __shared__ float Srow[WAVES_][16];
  if (lane == 0) {
#pragma unroll
    for (int r = 0; r < 8; ++r) Srow[wave][r] = c[r];
  }
  if (lane == 16) {
#pragma unroll
    for (int r = 0; r < 8; ++r) Srow[wave][8 + r] = c[r];
  }
  __syncthreads();
  const float S = Srow[wave][M];

  const float pl = mx + __logf(S) - dot;   // pn_loss (cross entropy)

  // ---- per-row outputs on the low lane of each pair ----
  const size_t row = (size_t)bIdx * D_ + (size_t)(valid ? d : 0);
  int np = 0; float ps = 0.0f; float lsum = 0.0f;
  if (valid && !isHigh) {
    const unsigned char p = pos[row];
    if (p) {
      np = 1; ps = pl;
      float acc = 0.0f;
#pragma unroll
      for (int j = 0; j < 4; ++j) {
        const float df = lp[M * STRIDE_ + j] - gt_loc[row * 4 + j];
        const float ad = fabsf(df);
        acc += (ad < 1.0f) ? 0.5f * df * df : (ad - 0.5f);
      }
      lsum = acc;
      nloss[row] = NEG_FILL_;
    } else {
      nloss[row] = pl;
    }
  }

  // ---- deterministic block reduction of (Npos, p_sum, loc_sum) ----
#pragma unroll
  for (int off = 16; off > 0; off >>= 1) {
    np   += __shfl_down(np, off, 32);
    ps   += __shfl_down(ps, off, 32);
    lsum += __shfl_down(lsum, off, 32);
  }
  __shared__ int   wN[WAVES_];
  __shared__ float wP[WAVES_], wL[WAVES_];
  if (lane == 0) { wN[wave] = np; wP[wave] = ps; wL[wave] = lsum; }
  __syncthreads();
  if (tid == 0) {
    int tN = 0; float tP = 0.0f, tL = 0.0f;
#pragma unroll
    for (int w = 0; w < WAVES_; ++w) { tN += wN[w]; tP += wP[w]; tL += wL[w]; }
    const int part = bIdx * NCHUNK + chnk;
    partN[part] = tN; partP[part] = tP; partLoc[part] = tL;
  }
}

// ---------------------------------------------------------------------------
// Kernel 2: reduce per-block partials -> per-batch Npos / p_sum / loc_sum
// ---------------------------------------------------------------------------
__global__ void __launch_bounds__(256) ssd_reduce_partials(
    const float* __restrict__ partP, const float* __restrict__ partLoc,
    const int* __restrict__ partN,
    float* __restrict__ Psum, float* __restrict__ LocSum, int* __restrict__ Npos)
{
  const int b = blockIdx.x;
  const int tid = threadIdx.x;
  int n = 0; float p = 0.0f, l = 0.0f;
  for (int i = tid; i < NCHUNK; i += 256) {
    const int idx = b * NCHUNK + i;
    n += partN[idx]; p += partP[idx]; l += partLoc[idx];
  }
  __shared__ int   rN[256];
  __shared__ float rP[256], rL[256];
  rN[tid] = n; rP[tid] = p; rL[tid] = l;
  __syncthreads();
  for (int s = 128; s > 0; s >>= 1) {
    if (tid < s) { rN[tid] += rN[tid + s]; rP[tid] += rP[tid + s]; rL[tid] += rL[tid + s]; }
    __syncthreads();
  }
  if (tid == 0) { Npos[b] = rN[0]; Psum[b] = rP[0]; LocSum[b] = rL[0]; }
}

// ---------------------------------------------------------------------------
// Kernel 3: exact top-k sum per batch via 32-pass radix select (no sort)
// ---------------------------------------------------------------------------
__global__ void __launch_bounds__(256) ssd_topk_kernel(
    const float* __restrict__ nloss, const int* __restrict__ Npos,
    float* __restrict__ Nsum)
{
  const int b   = blockIdx.x;
  const int tid = threadIdx.x;
  const int np  = Npos[b];
  long long k = (long long)(D_ - np);
  const long long k2 = 3LL * (long long)np;
  if (k2 < k) k = k2;
  if (np <= 0 || k <= 0) { if (tid == 0) Nsum[b] = 0.0f; return; }

  const float* nl = nloss + (size_t)b * D_;
  __shared__ int   redI[256];
  __shared__ float redF[256];

  unsigned prefix = 0;
  long long kk = k;
  for (int bit = 31; bit >= 0; --bit) {
    const unsigned cand = prefix | (1u << bit);
    int c = 0;
    for (int i = tid; i < D_; i += 256) {
      const unsigned u = mapF(nl[i]);
      if ((u >> bit) == (cand >> bit)) c++;
    }
    redI[tid] = c; __syncthreads();
    for (int s = 128; s > 0; s >>= 1) { if (tid < s) redI[tid] += redI[tid + s]; __syncthreads(); }
    const long long cnt = redI[0]; __syncthreads();
    if (cnt >= kk) prefix = cand; else kk -= cnt;
  }

  // sum strictly-greater elements, handle ties at the k-th value exactly
  int cg = 0; float sg = 0.0f;
  for (int i = tid; i < D_; i += 256) {
    const float x = nl[i];
    if (mapF(x) > prefix) { cg++; sg += x; }
  }
  redI[tid] = cg; redF[tid] = sg; __syncthreads();
  for (int s = 128; s > 0; s >>= 1) {
    if (tid < s) { redI[tid] += redI[tid + s]; redF[tid] += redF[tid + s]; }
    __syncthreads();
  }
  if (tid == 0) Nsum[b] = redF[0] + (float)(k - (long long)redI[0]) * unmapF(prefix);
}

// ---------------------------------------------------------------------------
// Kernel 4: combine per-batch losses, mean -> scalar
// ---------------------------------------------------------------------------
__global__ void __launch_bounds__(64) ssd_final_kernel(
    const float* __restrict__ Psum, const float* __restrict__ LocSum,
    const float* __restrict__ Nsum, const int* __restrict__ Npos,
    float* __restrict__ out)
{
  __shared__ float red[64];
  const int b = threadIdx.x;
  float v = 0.0f;
  const int np = Npos[b];
  if (np > 0) {
    const float ns = (float)np;
    v = (Psum[b] + Nsum[b]) / ns + LocSum[b] / ns;   // ALPHA = 1
  }
  red[b] = v;
  __syncthreads();
  for (int s = 32; s > 0; s >>= 1) { if (b < s) red[b] += red[b + s]; __syncthreads(); }
  if (b == 0) out[0] = red[0] / (float)B_;
}

// ---------------------------------------------------------------------------
extern "C" void kernel_launch(void* const* d_in, const int* in_sizes, int n_in,
                              void* d_out, int out_size, void* d_ws, size_t ws_size,
                              hipStream_t stream) {
  (void)in_sizes; (void)n_in; (void)out_size; (void)ws_size;
  const float*         predicts = (const float*)d_in[0];
  const float*         gt_loc   = (const float*)d_in[1];
  const float*         gt_conf  = (const float*)d_in[2];
  const unsigned char* pos      = (const unsigned char*)d_in[3];  // jax bool = 1 byte

  float* nloss   = (float*)d_ws;                    // B*D
  float* partP   = nloss + (size_t)B_ * D_;         // NPART
  float* partLoc = partP + NPART;                   // NPART
  int*   partN   = (int*)(partLoc + NPART);         // NPART
  float* Psum    = (float*)(partN + NPART);         // B
  float* LocSum  = Psum + B_;                       // B
  float* Nsum    = LocSum + B_;                     // B
  int*   Npos    = (int*)(Nsum + B_);               // B

  dim3 g1(NCHUNK, B_);
  ssd_row_kernel<<<g1, 128, 0, stream>>>(predicts, gt_loc, gt_conf, pos,
                                         nloss, partP, partLoc, partN);
  ssd_reduce_partials<<<B_, 256, 0, stream>>>(partP, partLoc, partN,
                                              Psum, LocSum, Npos);
  ssd_topk_kernel<<<B_, 256, 0, stream>>>(nloss, Npos, Nsum);
  ssd_final_kernel<<<1, 64, 0, stream>>>(Psum, LocSum, Nsum, Npos, (float*)d_out);
}
